// MambaSSM_6648609374651
// MI455X (gfx1250) — matrix operand order repaired and checked by reading, not compiled, not run on hardware
//
#include <hip/hip_runtime.h>
#include <hip/hip_bf16.h>
#include <hip/hip_fp16.h>

// Mamba selective-scan forward for MI455X (gfx1250, wave32, WMMA).
// Dims fixed by the reference: B=2, L=D=S=R=384, conv K=4.

#define BB 2
#define LL 384
#define DD 384
#define SS 384
#define KK 384          // GEMM reduction dim
#define MTOT (BB * LL)  // 768 GEMM rows

typedef __attribute__((ext_vector_type(16))) _Float16 v16h;
typedef __attribute__((ext_vector_type(8)))  _Float16 v8h;
typedef __attribute__((ext_vector_type(4)))  _Float16 v4h;
typedef __attribute__((ext_vector_type(8)))  float    v8f;

// ---------------------------------------------------------------------------
// f32 -> f16 pre-conversion (vectorized: b128 in, b64 out). Done once per
// tensor so the GEMM inner loop is pure loads + v_wmma.
// ---------------------------------------------------------------------------
__global__ void f32_to_f16_kernel(const float* __restrict__ in,
                                  _Float16* __restrict__ out, int n4) {
    const int i = blockIdx.x * blockDim.x + threadIdx.x;
    if (i >= n4) return;
    const float4 v = ((const float4*)in)[i];
    v4h o;
    o[0] = (_Float16)v.x; o[1] = (_Float16)v.y;
    o[2] = (_Float16)v.z; o[3] = (_Float16)v.w;
    ((v4h*)out)[i] = o;
}

// ---------------------------------------------------------------------------
// WMMA GEMM: C[M,N] = act(A[M,K] @ W[N,K]^T + bias), A and W pre-converted f16.
// One wave computes a 16x32 tile: the A fragment is reused by two back-to-back
// v_wmma_f32_16x16x32_f16 issues per K-step (independent accumulators -> no
// RAW hazard, fills the XDL pipe).
//
// ISA VGPR layouts (CDNA5 ISA):
//   A 16x32 f16 : lane: m = lane%16, g = lane/16.
//                 halves j=0..7  -> k = 8g + j        (contiguous 8 = b128)
//                 halves j=8..15 -> k = 16 + 8g + j-8 (contiguous 8 = b128)
//   B 32x16 f16 : lane: n = lane%16, this code, A and B alike: halves j=0..15 -> k = 16g + j (2x b128)
//   C/D 16x16 f32: VGPR i, lane: m = i + 8*(lane/16), n = lane%16
// ACT: 0 = f32 store, 1 = softplus(z + bias[n]) f32 store, 2 = f16 store
// ---------------------------------------------------------------------------
template <int ACT>
__global__ __launch_bounds__(256)
void wmma_gemm_f16nt(const _Float16* __restrict__ A,
                     const _Float16* __restrict__ W,
                     const float* __restrict__ bias,
                     float* __restrict__ Cf,
                     _Float16* __restrict__ Ch,
                     int M, int N, int K) {
    const int lane = threadIdx.x & 31;
    const int wave = threadIdx.x >> 5;
    const int tilesN2 = N >> 5;                 // 32-wide N tiles
    const int numTiles = (M >> 4) * tilesN2;
    const int tile = blockIdx.x * (blockDim.x >> 5) + wave;
    if (tile >= numTiles) return;

    const int m0 = (tile / tilesN2) << 4;
    const int n0 = (tile % tilesN2) << 5;
    const int half = lane & 15;   // m for A-frag, n for B-frags
    const int g    = lane >> 4;   // K-half selector

    const _Float16* __restrict__ arow  = A + (size_t)(m0 + half) * K;
    const _Float16* __restrict__ wrow0 = W + (size_t)(n0 + half) * K;
    const _Float16* __restrict__ wrow1 = W + (size_t)(n0 + 16 + half) * K;

    v8f acc0 = {};
    v8f acc1 = {};
    for (int k0 = 0; k0 < K; k0 += 32) {
        const v8h aLo = *(const v8h*)(arow + k0 + 16 * g);
        const v8h aHi = *(const v8h*)(arow + k0 + 16 * g + 8);
        const v8h b0a = *(const v8h*)(wrow0 + k0 + 16 * g);
        const v8h b0b = *(const v8h*)(wrow0 + k0 + 16 * g + 8);
        const v8h b1a = *(const v8h*)(wrow1 + k0 + 16 * g);
        const v8h b1b = *(const v8h*)(wrow1 + k0 + 16 * g + 8);

        const v16h af = __builtin_shufflevector(aLo, aHi,
            0,1,2,3,4,5,6,7,8,9,10,11,12,13,14,15);
        const v16h bf0 = __builtin_shufflevector(b0a, b0b,
            0,1,2,3,4,5,6,7,8,9,10,11,12,13,14,15);
        const v16h bf1 = __builtin_shufflevector(b1a, b1b,
            0,1,2,3,4,5,6,7,8,9,10,11,12,13,14,15);

        acc0 = __builtin_amdgcn_wmma_f32_16x16x32_f16(
            false, af, false, bf0, (short)0, acc0, false, false);
        acc1 = __builtin_amdgcn_wmma_f32_16x16x32_f16(
            false, af, false, bf1, (short)0, acc1, false, false);
    }

#pragma unroll
    for (int nt = 0; nt < 2; ++nt) {
        const v8f acc = nt ? acc1 : acc0;
        const int n = n0 + 16 * nt + half;
        float bv = 0.f;
        if (ACT == 1) bv = bias[n];
#pragma unroll
        for (int i = 0; i < 8; ++i) {
            const int m = m0 + i + 8 * g;
            float v = acc[i];
            if (ACT == 1) {
                v += bv;                                      // + dt_b
                v = (v > 20.f) ? v : __logf(1.f + __expf(v)); // softplus (fast)
            }
            if (ACT == 2) Ch[(size_t)m * N + n] = (_Float16)v;
            else          Cf[(size_t)m * N + n] = v;
        }
    }
}

// ---------------------------------------------------------------------------
// Causal depthwise conv1d (K=4, left pad 3) + bias + SiLU -> u (b, L, d)
// ---------------------------------------------------------------------------
__global__ void conv_silu_kernel(const float* __restrict__ x,
                                 const float* __restrict__ cw,
                                 const float* __restrict__ cb,
                                 float* __restrict__ u) {
    const int idx = blockIdx.x * blockDim.x + threadIdx.x;
    if (idx >= BB * LL * DD) return;
    const int d = idx % DD;
    const int l = (idx / DD) % LL;
    const int b = idx / (DD * LL);
    const float* xb = x + (size_t)b * LL * DD + d;
    float acc = cb[d];
#pragma unroll
    for (int k = 0; k < 4; ++k) {
        const int ls = l - 3 + k;
        if (ls >= 0) acc += xb[(size_t)ls * DD] * cw[d * 4 + k];
    }
    u[idx] = acc / (1.f + __expf(-acc));   // SiLU
}

// ---------------------------------------------------------------------------
// Selective scan. One block per (batch, 16 d-rows): 48 blocks x 256 threads.
// Thread t: d_local = t/16 (16 consecutive threads = half-wave share one d),
//           owns s = (t%16) + 16*j for j=0..23. h and -exp(A_log) live in
//           registers for all 384 steps; Bm/Cm rows staged into LDS per step
//           with b128 transfers; next step's rows prefetched into WGP$
//           (global_prefetch_b8) while this step computes.
// y-dot reduced across the 16-lane half-wave with __shfl_xor(width=16).
// ---------------------------------------------------------------------------
#define ROWS 16
#define SCAN_THREADS 256
#define SCHUNK (SS / 16)   // 24

__global__ __launch_bounds__(SCAN_THREADS)
void scan_kernel(const float* __restrict__ dt,
                 const float* __restrict__ u,
                 const float* __restrict__ Bm,
                 const float* __restrict__ Cm,
                 const float* __restrict__ A_log,
                 const float* __restrict__ Dv,
                 float* __restrict__ y) {
    __shared__ __align__(16) float sB[SS];
    __shared__ __align__(16) float sC[SS];
    __shared__ float sdt[ROWS];
    __shared__ float su[ROWS];

    const int b  = blockIdx.x / (DD / ROWS);
    const int d0 = (blockIdx.x % (DD / ROWS)) * ROWS;
    const int t  = threadIdx.x;
    const int dl = t >> 4;     // d_local in [0,16)
    const int sl = t & 15;     // s base
    const int d  = d0 + dl;

    float h[SCHUNK], Aval[SCHUNK];
    const float Dd = Dv[d];
#pragma unroll
    for (int j = 0; j < SCHUNK; ++j) {
        h[j] = 0.f;
        Aval[j] = -__expf(A_log[(size_t)d * SS + sl + 16 * j]);  // A = -exp(A_log)
    }

    const float* dtb = dt + (size_t)b * LL * DD;
    const float* ub  = u  + (size_t)b * LL * DD;
    const float* Bb  = Bm + (size_t)b * LL * SS;
    const float* Cb  = Cm + (size_t)b * LL * SS;
    float*       yb  = y  + (size_t)b * LL * DD;

    for (int ts = 0; ts < LL; ++ts) {
        // stage this step's operands into LDS (b128 transfers)
        if (t < ROWS) {
            sdt[t] = dtb[(size_t)ts * DD + d0 + t];
            su[t]  = ub [(size_t)ts * DD + d0 + t];
        }
        if (t < 96) {
            ((float4*)sB)[t] = ((const float4*)(Bb + (size_t)ts * SS))[t];
        } else if (t < 192) {
            ((float4*)sC)[t - 96] = ((const float4*)(Cb + (size_t)ts * SS))[t - 96];
        }
        // prefetch next step's rows into near cache while we compute
        if (ts + 1 < LL) {
            if (t < 24)       __builtin_prefetch(Bb + (size_t)(ts + 1) * SS + t * 16, 0, 0);
            else if (t < 48)  __builtin_prefetch(Cb + (size_t)(ts + 1) * SS + (t - 24) * 16, 0, 0);
        }
        __syncthreads();

        const float dtv = sdt[dl];
        const float uv  = su[dl];
        const float du  = dtv * uv;
        float part = 0.f;
#pragma unroll
        for (int j = 0; j < SCHUNK; ++j) {
            const int s = sl + 16 * j;
            const float dA = __expf(dtv * Aval[j]);
            h[j] = dA * h[j] + du * sB[s];
            part += h[j] * sC[s];
        }
        // reduce over the 16-lane half-wave (wave32: groups of 16)
        part += __shfl_xor(part, 1, 16);
        part += __shfl_xor(part, 2, 16);
        part += __shfl_xor(part, 4, 16);
        part += __shfl_xor(part, 8, 16);
        if (sl == 0) yb[(size_t)ts * DD + d] = part + uv * Dd;
        __syncthreads();
    }
}

// ---------------------------------------------------------------------------
// Launch. Inputs (setup_inputs order):
//  0:x 1:A_log 2:D 3:dt_w 4:dt_b 5:B_w 6:C_w 7:conv_w 8:conv_b
// Workspace: f32 slices dt|Bm|Cm|u (768*384 each), then f16 xh|xdh (768*384)
// and wdt|wB|wC (384*384).  Total ~6.8 MB.
// ---------------------------------------------------------------------------
extern "C" void kernel_launch(void* const* d_in, const int* in_sizes, int n_in,
                              void* d_out, int out_size, void* d_ws, size_t ws_size,
                              hipStream_t stream) {
    const float* x      = (const float*)d_in[0];
    const float* A_log  = (const float*)d_in[1];
    const float* Dvec   = (const float*)d_in[2];
    const float* dt_w   = (const float*)d_in[3];
    const float* dt_b   = (const float*)d_in[4];
    const float* B_w    = (const float*)d_in[5];
    const float* C_w    = (const float*)d_in[6];
    const float* conv_w = (const float*)d_in[7];
    const float* conv_b = (const float*)d_in[8];
    float* y = (float*)d_out;

    const size_t slice  = (size_t)MTOT * DD;   // 294912 elements
    const size_t wslice = (size_t)DD * KK;     // 147456 elements
    float* dt = (float*)d_ws;
    float* Bm = dt + slice;
    float* Cm = Bm + slice;
    float* u  = Cm + slice;
    _Float16* xh  = (_Float16*)(u + slice);
    _Float16* xdh = xh + slice;                // f16 x_dbl (GEMM output)
    _Float16* wdt = xdh + slice;
    _Float16* wB  = wdt + wslice;
    _Float16* wC  = wB + wslice;

    // --- pre-convert GEMM operands to f16 (one pass each) ---
    const int cvtThreads = 256;
    f32_to_f16_kernel<<<(int)((slice / 4 + cvtThreads - 1) / cvtThreads), cvtThreads, 0, stream>>>(x, xh, (int)(slice / 4));
    f32_to_f16_kernel<<<(int)((wslice / 4 + cvtThreads - 1) / cvtThreads), cvtThreads, 0, stream>>>(dt_w, wdt, (int)(wslice / 4));
    f32_to_f16_kernel<<<(int)((wslice / 4 + cvtThreads - 1) / cvtThreads), cvtThreads, 0, stream>>>(B_w, wB, (int)(wslice / 4));
    f32_to_f16_kernel<<<(int)((wslice / 4 + cvtThreads - 1) / cvtThreads), cvtThreads, 0, stream>>>(C_w, wC, (int)(wslice / 4));

    // --- WMMA GEMMs: (768/16)*(384/32) = 576 wave-tiles, 8 waves/block ---
    const int tiles = (MTOT / 16) * (DD / 32);
    const int gemmBlocks = (tiles + 7) / 8;

    // dt = softplus(x @ dt_w^T + dt_b)
    wmma_gemm_f16nt<1><<<gemmBlocks, 256, 0, stream>>>(xh, wdt, dt_b, dt, nullptr, MTOT, DD, KK);
    // x_dbl = x @ B_w^T   (stored f16: consumed only as a GEMM A-operand)
    wmma_gemm_f16nt<2><<<gemmBlocks, 256, 0, stream>>>(xh, wB, nullptr, nullptr, xdh, MTOT, SS, KK);
    // Bm = x_dbl @ B_w^T ; Cm = x_dbl @ C_w^T
    wmma_gemm_f16nt<0><<<gemmBlocks, 256, 0, stream>>>(xdh, wB, nullptr, Bm, nullptr, MTOT, SS, KK);
    wmma_gemm_f16nt<0><<<gemmBlocks, 256, 0, stream>>>(xdh, wC, nullptr, Cm, nullptr, MTOT, SS, KK);

    // u = silu(conv1d(x) + conv_b)
    const int n = BB * LL * DD;
    conv_silu_kernel<<<(n + 255) / 256, 256, 0, stream>>>(x, conv_w, conv_b, u);

    // selective scan -> y
    scan_kernel<<<BB * (DD / ROWS), SCAN_THREADS, 0, stream>>>(
        dt, u, Bm, Cm, A_log, Dvec, y);
}
